// LSTMNet_87497073754531
// MI455X (gfx1250) — compile-verified, matrix operand order checked by reading, not run on hardware
//
#include <hip/hip_runtime.h>
#include <hip/hip_bf16.h>
#include <math.h>

typedef __attribute__((ext_vector_type(16))) _Float16 v16h;
typedef __attribute__((ext_vector_type(8)))  _Float16 v8h;
typedef __attribute__((ext_vector_type(8)))  float    v8f;

#define B_SZ   4096
#define T_SZ   60
#define D_SZ   200
#define H_SZ   64
#define G_SZ   256          // 4H
#define DPAD   224          // 200 padded to 7*32
#define M_SZ   (B_SZ * T_SZ)   // 245760
#define FCK    7680         // 2*T*H
#define FCN    128
#define OUTN   15

// ---------------------------------------------------------------------------
// WMMA helper: D = A(16x32 f16) * B(32x16 f16) + C(16x16 f32)
// ---------------------------------------------------------------------------
__device__ __forceinline__ v8f wmma16(v16h a, v16h b, v8f c) {
    return __builtin_amdgcn_wmma_f32_16x16x32_f16(
        /*neg_a=*/false, a, /*neg_b=*/false, b,
        /*c_mod=*/(short)0, c, /*reuse_a=*/false, /*reuse_b=*/false);
}

// Load one 16x32 fragment (A-matrix layout, also used for B=[N][K] rows).
// Lane r (0-15): row r0+r, K = k0+0..7 and k0+16..23
// Lane r+16   : row r0+r, K = k0+8..15 and k0+24..31
template <typename P>
__device__ __forceinline__ v16h load_frag16(P base, int ld, int r0, int k0, int lane) {
    int r  = r0 + (lane & 15);
    int kb = k0 + ((lane >> 4) << 3);
    P p = base + r * ld + kb;
    v8h lo = *reinterpret_cast<const v8h*>(p);
    v8h hi = *reinterpret_cast<const v8h*>(p + 16);
    v16h out;
#pragma unroll
    for (int i = 0; i < 8; ++i) { out[i] = lo[i]; out[i + 8] = hi[i]; }
    return out;
}

// ---------------------------------------------------------------------------
// Pack f32 [rows][K] -> f16 [rows][Kpad] (zero padded)
// ---------------------------------------------------------------------------
__global__ void pack_f16_kernel(const float* __restrict__ src, _Float16* __restrict__ dst,
                                int rows, int K, int Kpad) {
    int idx = blockIdx.x * blockDim.x + threadIdx.x;
    if (idx >= rows * Kpad) return;
    int r = idx / Kpad, k = idx - r * Kpad;
    dst[idx] = (k < K) ? (_Float16)src[r * K + k] : (_Float16)0.0f;
}

__global__ void add_bias_kernel(const float* a, const float* b, float* o, int n) {
    int i = blockIdx.x * blockDim.x + threadIdx.x;
    if (i < n) o[i] = a[i] + b[i];
}

// concat + relu -> fp16 FC input  [B][7680]
__global__ void pack_fcin_kernel(const _Float16* __restrict__ h1a,
                                 const _Float16* __restrict__ h1b,
                                 _Float16* __restrict__ fcin) {
    int idx = blockIdx.x * blockDim.x + threadIdx.x;
    if (idx >= B_SZ * FCK) return;
    int b = idx / FCK, k = idx - b * FCK;
    float v = (k < FCK / 2) ? (float)h1a[b * (FCK / 2) + k]
                            : (float)h1b[b * (FCK / 2) + (k - FCK / 2)];
    fcin[idx] = (_Float16)fmaxf(v, 0.0f);
}

// ---------------------------------------------------------------------------
// Generic WMMA GEMM: C[M][N] = A[M][K]f16 * W[N][K]f16^T + bias, opt ReLU
// block = 256 threads = 8 waves; wave tile 32x32; block tile 128(M) x 64(N)
// grid = (N/64, M/128)
// ---------------------------------------------------------------------------
__global__ void gemm_wmma_kernel(const _Float16* __restrict__ A, int lda,
                                 const _Float16* __restrict__ W, int ldb,
                                 const float* __restrict__ bias,
                                 float* __restrict__ C, int ldc,
                                 int K, int do_relu) {
    const int lane = threadIdx.x & 31;
    const int wave = threadIdx.x >> 5;
    const int m0 = blockIdx.y * 128 + (wave >> 1) * 32;
    const int n0 = blockIdx.x * 64  + (wave & 1) * 32;

    v8f c00 = {}, c01 = {}, c10 = {}, c11 = {};
    for (int k0 = 0; k0 < K; k0 += 32) {
        v16h a0 = load_frag16(A, lda, m0,      k0, lane);
        v16h a1 = load_frag16(A, lda, m0 + 16, k0, lane);
        v16h b0 = load_frag16(W, ldb, n0,      k0, lane);
        v16h b1 = load_frag16(W, ldb, n0 + 16, k0, lane);
        c00 = wmma16(a0, b0, c00);
        c01 = wmma16(a0, b1, c01);
        c10 = wmma16(a1, b0, c10);
        c11 = wmma16(a1, b1, c11);
    }

    const int r = lane & 15, hf = lane >> 4;
#pragma unroll
    for (int tm = 0; tm < 2; ++tm) {
#pragma unroll
        for (int tn = 0; tn < 2; ++tn) {
            v8f acc = (tm == 0) ? (tn == 0 ? c00 : c01) : (tn == 0 ? c10 : c11);
            int col = n0 + tn * 16 + r;
            float bv = bias ? bias[col] : 0.0f;
#pragma unroll
            for (int i = 0; i < 8; ++i) {
                int row = m0 + tm * 16 + hf * 8 + i;
                float v = acc[i] + bv;
                if (do_relu) v = fmaxf(v, 0.0f);
                C[row * ldc + col] = v;
            }
        }
    }
}

// ---------------------------------------------------------------------------
// LSTM recurrence: gates[t] = xg[:,t,:] + h @ Whh^T, then elementwise update.
// One block = 16 batch rows, 8 waves; wave w owns gate cols [32w, 32w+32).
// Whh fragments live in registers across all T steps; h (f16) + gates (f32)
// staged in LDS.  xg: [B*T][256] f32, rows m = b*T + t.  hout: [B*T][64] f16.
// ---------------------------------------------------------------------------
__global__ void lstm_rec_kernel(const float* __restrict__ xg,
                                const _Float16* __restrict__ Whh,  // [256][64] f16
                                _Float16* __restrict__ hout,
                                int Tlen) {
    __shared__ __align__(16) _Float16 hs[16][H_SZ];
    __shared__ float gs[16][G_SZ];

    const int tid  = threadIdx.x;
    const int lane = tid & 31;
    const int wave = tid >> 5;
    const int b0   = blockIdx.x * 16;
    const int n0   = wave * 32;
    const int r    = lane & 15, hf = lane >> 4;

    // Preload W_hh fragments (K=64 -> 2 k-steps, 2 n-tiles) : 4 frags resident
    v16h w00 = load_frag16(Whh, H_SZ, n0,      0,  lane);
    v16h w10 = load_frag16(Whh, H_SZ, n0,      32, lane);
    v16h w01 = load_frag16(Whh, H_SZ, n0 + 16, 0,  lane);
    v16h w11 = load_frag16(Whh, H_SZ, n0 + 16, 32, lane);

    // zero h in LDS; c-state in registers (each thread owns 4 cells)
    {
        _Float16* hp = &hs[0][0];
#pragma unroll
        for (int i = 0; i < 4; ++i) hp[tid * 4 + i] = (_Float16)0.0f;
    }
    const int u = tid & 63;            // hidden index
    const int rbase = (tid >> 6) * 4;  // 4 batch rows per thread
    float cst[4] = {0.f, 0.f, 0.f, 0.f};
    __syncthreads();

    for (int t = 0; t < Tlen; ++t) {
        // ---- gates = xg + h @ Whh^T -------------------------------------
        v8f acc0, acc1;
#pragma unroll
        for (int i = 0; i < 8; ++i) {
            int row = hf * 8 + i;
            int m = (b0 + row) * Tlen + t;
            acc0[i] = xg[m * G_SZ + n0 + r];
            acc1[i] = xg[m * G_SZ + n0 + 16 + r];
        }
        v16h a0 = load_frag16((const _Float16*)&hs[0][0], H_SZ, 0, 0,  lane);
        v16h a1 = load_frag16((const _Float16*)&hs[0][0], H_SZ, 0, 32, lane);
        acc0 = wmma16(a0, w00, acc0);
        acc0 = wmma16(a1, w10, acc0);
        acc1 = wmma16(a0, w01, acc1);
        acc1 = wmma16(a1, w11, acc1);
#pragma unroll
        for (int i = 0; i < 8; ++i) {
            int row = hf * 8 + i;
            gs[row][n0 + r]      = acc0[i];
            gs[row][n0 + 16 + r] = acc1[i];
        }
        __syncthreads();

        // ---- elementwise: i,f,g,o -> c,h --------------------------------
#pragma unroll
        for (int rr = 0; rr < 4; ++rr) {
            int row = rbase + rr;
            float ig = gs[row][u];
            float fg = gs[row][H_SZ + u];
            float gg = gs[row][2 * H_SZ + u];
            float og = gs[row][3 * H_SZ + u];
            float si = 1.0f / (1.0f + __expf(-ig));
            float sf = 1.0f / (1.0f + __expf(-fg));
            float so = 1.0f / (1.0f + __expf(-og));
            float c  = sf * cst[rr] + si * tanhf(gg);
            cst[rr]  = c;
            float h  = so * tanhf(c);
            hs[row][u] = (_Float16)h;
            hout[((b0 + row) * Tlen + t) * H_SZ + u] = (_Float16)h;
        }
        __syncthreads();
    }
}

// ---------------------------------------------------------------------------
// FC2 + partial sum-of-squares (deterministic two-pass norm)
// ---------------------------------------------------------------------------
__global__ void fc2_partial_kernel(const float* __restrict__ fc1o,
                                   const float* __restrict__ w2,
                                   const float* __restrict__ b2,
                                   float* __restrict__ z,
                                   float* __restrict__ partial) {
    __shared__ float red[256];
    int idx = blockIdx.x * 256 + threadIdx.x;
    float acc = 0.0f, sq = 0.0f;
    if (idx < B_SZ * OUTN) {
        int b = idx / OUTN, j = idx - b * OUTN;
        acc = b2[j];
#pragma unroll 8
        for (int k = 0; k < FCN; ++k) acc += fc1o[b * FCN + k] * w2[j * FCN + k];
        z[idx] = acc;
        sq = acc * acc;
    }
    red[threadIdx.x] = sq;
    __syncthreads();
    for (int s = 128; s > 0; s >>= 1) {
        if (threadIdx.x < s) red[threadIdx.x] += red[threadIdx.x + s];
        __syncthreads();
    }
    if (threadIdx.x == 0) partial[blockIdx.x] = red[0];
}

__global__ void norm_reduce_kernel(const float* __restrict__ partial, int n,
                                   float* __restrict__ sumsq) {
    __shared__ float red[256];
    float v = (threadIdx.x < n) ? partial[threadIdx.x] : 0.0f;
    red[threadIdx.x] = v;
    __syncthreads();
    for (int s = 128; s > 0; s >>= 1) {
        if (threadIdx.x < s) red[threadIdx.x] += red[threadIdx.x + s];
        __syncthreads();
    }
    if (threadIdx.x == 0) sumsq[0] = red[0];
}

__global__ void scale_out_kernel(const float* __restrict__ z,
                                 const float* __restrict__ sumsq,
                                 float* __restrict__ out, int n) {
    int i = blockIdx.x * blockDim.x + threadIdx.x;
    if (i < n) out[i] = z[i] * rsqrtf(sumsq[0]);
}

// ---------------------------------------------------------------------------
// Host orchestration
// ---------------------------------------------------------------------------
extern "C" void kernel_launch(void* const* d_in, const int* in_sizes, int n_in,
                              void* d_out, int out_size, void* d_ws, size_t ws_size,
                              hipStream_t stream) {
    const float* x_in[2]  = { (const float*)d_in[0], (const float*)d_in[1] };
    const float* Wih0 = (const float*)d_in[2];
    const float* Whh0 = (const float*)d_in[3];
    const float* bih0 = (const float*)d_in[4];
    const float* bhh0 = (const float*)d_in[5];
    const float* Wih1 = (const float*)d_in[6];
    const float* Whh1 = (const float*)d_in[7];
    const float* bih1 = (const float*)d_in[8];
    const float* bhh1 = (const float*)d_in[9];
    const float* fc1w = (const float*)d_in[10];
    const float* fc1b = (const float*)d_in[11];
    const float* fc2w = (const float*)d_in[12];
    const float* fc2b = (const float*)d_in[13];
    float* out = (float*)d_out;

    // workspace layout (256B aligned slices)
    char* base = (char*)d_ws;
    size_t off = 0;
    auto alloc = [&](size_t bytes) -> char* {
        char* p = base + off;
        off += (bytes + 255) & ~(size_t)255;
        return p;
    };
    _Float16* xpack = (_Float16*)alloc((size_t)M_SZ * DPAD * 2);
    float*    xg    = (float*)   alloc((size_t)M_SZ * G_SZ * 4);
    _Float16* h0    = (_Float16*)alloc((size_t)M_SZ * H_SZ * 2);
    _Float16* h1a   = (_Float16*)alloc((size_t)M_SZ * H_SZ * 2);
    _Float16* h1b   = (_Float16*)alloc((size_t)M_SZ * H_SZ * 2);
    _Float16* fcin  = (_Float16*)alloc((size_t)B_SZ * FCK * 2);
    float*    fc1o  = (float*)   alloc((size_t)B_SZ * FCN * 4);
    float*    z     = (float*)   alloc((size_t)B_SZ * OUTN * 4);
    float*    part  = (float*)   alloc(256 * 4);
    float*    sumsq = (float*)   alloc(256);
    _Float16* wih0p = (_Float16*)alloc((size_t)G_SZ * DPAD * 2);
    _Float16* whh0p = (_Float16*)alloc((size_t)G_SZ * H_SZ * 2);
    _Float16* wih1p = (_Float16*)alloc((size_t)G_SZ * H_SZ * 2);
    _Float16* whh1p = (_Float16*)alloc((size_t)G_SZ * H_SZ * 2);
    _Float16* fc1wp = (_Float16*)alloc((size_t)FCN * FCK * 2);
    float*    bias0 = (float*)   alloc(G_SZ * 4);
    float*    bias1 = (float*)   alloc(G_SZ * 4);

    auto cdiv = [](long a, long b) { return (unsigned)((a + b - 1) / b); };

    // ---- pack weights / biases ------------------------------------------
    pack_f16_kernel<<<cdiv((long)G_SZ * DPAD, 256), 256, 0, stream>>>(Wih0, wih0p, G_SZ, D_SZ, DPAD);
    pack_f16_kernel<<<cdiv((long)G_SZ * H_SZ, 256), 256, 0, stream>>>(Whh0, whh0p, G_SZ, H_SZ, H_SZ);
    pack_f16_kernel<<<cdiv((long)G_SZ * H_SZ, 256), 256, 0, stream>>>(Wih1, wih1p, G_SZ, H_SZ, H_SZ);
    pack_f16_kernel<<<cdiv((long)G_SZ * H_SZ, 256), 256, 0, stream>>>(Whh1, whh1p, G_SZ, H_SZ, H_SZ);
    pack_f16_kernel<<<cdiv((long)FCN * FCK, 256), 256, 0, stream>>>(fc1w, fc1wp, FCN, FCK, FCK);
    add_bias_kernel<<<1, 256, 0, stream>>>(bih0, bhh0, bias0, G_SZ);
    add_bias_kernel<<<1, 256, 0, stream>>>(bih1, bhh1, bias1, G_SZ);

    // ---- LSTM stack for each input --------------------------------------
    for (int s = 0; s < 2; ++s) {
        _Float16* h1 = (s == 0) ? h1a : h1b;
        // x -> f16 padded [M][224]
        pack_f16_kernel<<<cdiv((long)M_SZ * DPAD, 256), 256, 0, stream>>>(
            x_in[s], xpack, M_SZ, D_SZ, DPAD);
        // xg0 = x @ Wih0^T + (b_ih0+b_hh0)
        gemm_wmma_kernel<<<dim3(G_SZ / 64, M_SZ / 128), 256, 0, stream>>>(
            xpack, DPAD, wih0p, DPAD, bias0, xg, G_SZ, DPAD, 0);
        // layer 0 recurrence
        lstm_rec_kernel<<<B_SZ / 16, 256, 0, stream>>>(xg, whh0p, h0, T_SZ);
        // xg1 = h0 @ Wih1^T + (b_ih1+b_hh1)
        gemm_wmma_kernel<<<dim3(G_SZ / 64, M_SZ / 128), 256, 0, stream>>>(
            h0, H_SZ, wih1p, H_SZ, bias1, xg, G_SZ, H_SZ, 0);
        // layer 1 recurrence
        lstm_rec_kernel<<<B_SZ / 16, 256, 0, stream>>>(xg, whh1p, h1, T_SZ);
    }

    // ---- head ------------------------------------------------------------
    pack_fcin_kernel<<<cdiv((long)B_SZ * FCK, 256), 256, 0, stream>>>(h1a, h1b, fcin);
    gemm_wmma_kernel<<<dim3(FCN / 64, B_SZ / 128), 256, 0, stream>>>(
        fcin, FCK, fc1wp, FCK, fc1b, fc1o, FCN, FCK, 1);

    const int nz = B_SZ * OUTN;                  // 61440
    fc2_partial_kernel<<<cdiv(nz, 256), 256, 0, stream>>>(fc1o, fc2w, fc2b, z, part);
    norm_reduce_kernel<<<1, 256, 0, stream>>>(part, cdiv(nz, 256), sumsq);
    scale_out_kernel<<<cdiv(nz, 256), 256, 0, stream>>>(z, sumsq, out, nz);
}
